// TransformerBlock_240518169078
// MI455X (gfx1250) — compile-verified, matrix operand order checked by reading, not run on hardware
//
#include <hip/hip_runtime.h>

typedef __attribute__((ext_vector_type(16))) __bf16 v16bf;
typedef __attribute__((ext_vector_type(8)))  __bf16 v8bf;
typedef __attribute__((ext_vector_type(8)))  float  v8f;

typedef unsigned int u32;
typedef u32 u32x4 __attribute__((ext_vector_type(4)));
typedef int i32x4 __attribute__((ext_vector_type(4)));
typedef int i32x8 __attribute__((ext_vector_type(8)));

#define DM   1024
#define DFF  4096
#define NH   16
#define DKH  64
#define BB   2
#define SSQ  2048
#define MTOK (BB * SSQ)

#if defined(__has_builtin)
#if __has_builtin(__builtin_amdgcn_tensor_load_to_lds)
#define HAS_TDM 1
#endif
#endif
#ifndef HAS_TDM
#define HAS_TDM 0
#endif

// ---------------------------------------------------------------------------
// Fragment load: 16x32 bf16 tile (row-major, leading dim ld) into WMMA A/B
// register layout. Lanes 0-15 hold row=lane, K = {0..7, 16..23}; lanes 16-31
// hold row=lane-16, K = {8..15, 24..31}. Two 16B vector loads per lane.
// Works for both global and LDS source (addrspace inferred after inlining).
// ---------------------------------------------------------------------------
__device__ __forceinline__ v16bf load_frag(const __bf16* base, int ld) {
  const int lane = threadIdx.x & 31;
  const int row  = lane & 15;
  const int kb   = (lane >> 4) << 3;
  const __bf16* p = base + (size_t)row * (size_t)ld + kb;
  v8bf lo = *(const v8bf*)(p);
  v8bf hi = *(const v8bf*)(p + 16);
  v16bf f;
#pragma unroll
  for (int i = 0; i < 8; ++i) { f[i] = lo[i]; f[i + 8] = hi[i]; }
  return f;
}

__device__ __forceinline__ v8f wmma_bf16(v16bf a, v16bf b, v8f c) {
  return __builtin_amdgcn_wmma_f32_16x16x32_bf16(false, a, false, b, (short)0, c,
                                                 false, false);
}

// ---------------------------------------------------------------------------
// TDM: issue a 2D tensor_load_to_lds of a bf16 tile (tile_d0 elems along K,
// tile_d1 rows) from global (row stride = stride0 elems) into LDS.
// D# group layouts per CDNA5 ISA 8.3/8.4; data_size=1 (2 bytes), count=1,
// type=2 ("image"), no multicast / iterate / padding.
// ---------------------------------------------------------------------------
#if HAS_TDM
__device__ __forceinline__ void tdm_load_2d(u32 lds_off, const void* gptr,
                                            u32 tensor_d0, u32 tensor_d1,
                                            u32 tile_d0, u32 tile_d1,
                                            u32 stride0) {
  unsigned long long ga = (unsigned long long)gptr;
  u32x4 g0;
  g0[0] = 1u;                                       // count=1 (valid user D#)
  g0[1] = lds_off;                                  // lds_addr (bytes)
  g0[2] = (u32)ga;                                  // global_addr[31:0]
  g0[3] = (u32)((ga >> 32) & 0x1FFFFFFu) | (2u << 30); // addr[56:32] | type=2
  i32x8 g1;
  g1[0] = (int)(1u << 16);                          // data_size=1 (2B), mask=0
  g1[1] = (int)((tensor_d0 & 0xFFFFu) << 16);       // tensor_dim0[15:0]
  g1[2] = (int)((tensor_d0 >> 16) | ((tensor_d1 & 0xFFFFu) << 16));
  g1[3] = (int)((tensor_d1 >> 16) | (tile_d0 << 16));  // dim1 hi | tile_dim0
  g1[4] = (int)tile_d1;                             // tile_dim1 (tile_dim2=0)
  g1[5] = (int)stride0;                             // tensor_dim0_stride[31:0]
  g1[6] = (int)((stride0 >> 16) >> 16);             // stride0 hi16 | stride1 lo=0
  g1[7] = 0;
  i32x4 z4 = {};
#if defined(__clang_major__) && (__clang_major__ >= 23)
  i32x8 z8 = {};
  __builtin_amdgcn_tensor_load_to_lds(g0, g1, z4, z4, z8, 0);
#else
  __builtin_amdgcn_tensor_load_to_lds(g0, g1, z4, z4, 0);
#endif
}
#endif

// ---------------------------------------------------------------------------
// fp32 -> bf16 conversion (grid-stride)
// ---------------------------------------------------------------------------
__global__ void cvt_f32_bf16(const float* __restrict__ in, __bf16* __restrict__ out,
                             int n) {
  int i = blockIdx.x * blockDim.x + threadIdx.x;
  int stride = gridDim.x * blockDim.x;
  for (; i < n; i += stride) out[i] = (__bf16)in[i];
}

// ---------------------------------------------------------------------------
// V transpose: [tok, H*DKH] token-major -> [B*H, DKH, S] (contig along S)
// ---------------------------------------------------------------------------
__global__ void transpose_v(const __bf16* __restrict__ v, __bf16* __restrict__ vt) {
  int i = blockIdx.x * blockDim.x + threadIdx.x;
  const int total = MTOK * DM;
  for (; i < total; i += gridDim.x * blockDim.x) {
    int tok = i / DM, c = i % DM;
    int b = tok / SSQ, s = tok % SSQ;
    int h = c / DKH,  d = c % DKH;
    vt[(((size_t)(b * NH + h) * DKH) + d) * SSQ + s] = v[i];
  }
}

// ---------------------------------------------------------------------------
// GEMM: out[M,N] = A[M,K](bf16) * W[N,K](bf16)^T + bias (+ReLU)
// block = 256 threads = 8 waves, tile 128(M) x 128(N); wave tile 32 x 64.
// Weight tile (128 x 64K, 16KB) staged into double-buffered LDS by the
// Tensor Data Mover; TDM of tile k+1 overlaps WMMA on tile k. All four
// B fragments of a 32-K step are preloaded so ds_load waits stagger
// against WMMA execution instead of serializing.
// ---------------------------------------------------------------------------
__global__ void gemm_wmma(const __bf16* __restrict__ A, const __bf16* __restrict__ W,
                          const float* __restrict__ bias,
                          float* __restrict__ outF, __bf16* __restrict__ outB,
                          int Ndim, int Kdim, int relu) {
  __shared__ __bf16 Bs[2][128 * 64];

  const int wave = threadIdx.x >> 5;
  const int lane = threadIdx.x & 31;
  const int wm = wave >> 1;                 // 0..3 -> M sub-tile
  const int wn = wave & 1;                  // 0..1 -> N half
  const int tm  = blockIdx.y * 128 + wm * 32;
  const int tnb = blockIdx.x * 128;

  v8f acc[2][4] = {};

  const __bf16* Wtile = W + (size_t)tnb * (size_t)Kdim;
  const __bf16* Arow0 = A + (size_t)tm * (size_t)Kdim;
  const __bf16* Arow1 = A + (size_t)(tm + 16) * (size_t)Kdim;

#if HAS_TDM
  if (wave == 0)
    tdm_load_2d((u32)(unsigned long long)&Bs[0][0], Wtile,
                (u32)Kdim, 128u, 64u, 128u, (u32)Kdim);
#endif

  int buf = 0;
  for (int kk = 0; kk < Kdim; kk += 64) {
#if HAS_TDM
    if (wave == 0) __builtin_amdgcn_s_wait_tensorcnt(0);
    __syncthreads();                       // tile kk visible to all waves
    if (wave == 0 && kk + 64 < Kdim)       // overlap DMA of next tile
      tdm_load_2d((u32)(unsigned long long)&Bs[buf ^ 1][0], Wtile + (kk + 64),
                  (u32)(Kdim - (kk + 64)), 128u, 64u, 128u, (u32)Kdim);
#else
    __syncthreads();
    for (int i = threadIdx.x; i < 128 * 8; i += 256) {   // 128 rows x 64 elems
      int r = i >> 3, c8 = (i & 7) << 3;
      *(v8bf*)&Bs[buf][r * 64 + c8] =
          *(const v8bf*)&Wtile[(size_t)r * Kdim + kk + c8];
    }
    __syncthreads();
#endif
    if (kk + 64 < Kdim) __builtin_prefetch(Arow0 + kk + 64, 0, 0);

    const __bf16* Bbase = &Bs[buf][(size_t)wn * 64 * 64];
#pragma unroll
    for (int ks = 0; ks < 64; ks += 32) {
      // preload all operand fragments first: one ds clause + one vmem clause,
      // waits stagger against the 8 WMMAs below
      v16bf bfr[4];
#pragma unroll
      for (int t = 0; t < 4; ++t)
        bfr[t] = load_frag(Bbase + (size_t)(t * 16) * 64 + ks, 64);
      v16bf a0 = load_frag(Arow0 + kk + ks, Kdim);
      v16bf a1 = load_frag(Arow1 + kk + ks, Kdim);
#pragma unroll
      for (int t = 0; t < 4; ++t) {
        acc[0][t] = wmma_bf16(a0, bfr[t], acc[0][t]);
        acc[1][t] = wmma_bf16(a1, bfr[t], acc[1][t]);
      }
    }
    buf ^= 1;
  }

  const int col = lane & 15;
  const int rb  = (lane >> 4) << 3;
#pragma unroll
  for (int s = 0; s < 2; ++s)
#pragma unroll
    for (int t = 0; t < 4; ++t) {
      int n = tnb + wn * 64 + t * 16 + col;
      float bv = bias ? bias[n] : 0.f;
#pragma unroll
      for (int r = 0; r < 8; ++r) {
        float vv = acc[s][t][r] + bv;
        if (relu) vv = fmaxf(vv, 0.f);
        size_t idx = (size_t)(tm + s * 16 + rb + r) * (size_t)Ndim + n;
        if (outF) outF[idx] = vv;
        if (outB) outB[idx] = (__bf16)vv;
      }
    }
}

// ---------------------------------------------------------------------------
// Flash attention: one wave (32 threads) per (b*h, 16-query tile).
// Online softmax with width-16 shfl reductions; P re-laid-out via LDS.
// ---------------------------------------------------------------------------
__global__ void attn_flash(const __bf16* __restrict__ Qb, const __bf16* __restrict__ Kb,
                           const __bf16* __restrict__ Vt, __bf16* __restrict__ Ob) {
  const int bh = blockIdx.y;            // 0..B*NH-1
  const int b  = bh >> 4, h = bh & 15;
  const int q0 = blockIdx.x * 16;
  const int lane = threadIdx.x & 31;
  const int col  = lane & 15;
  const int rb   = (lane >> 4) << 3;

  __shared__ __bf16 Plds[16 * 32];

  const __bf16* Qp = Qb + (size_t)(b * SSQ + q0) * DM + h * DKH;
  v16bf qa0 = load_frag(Qp, DM);        // d = 0..31
  v16bf qa1 = load_frag(Qp + 32, DM);   // d = 32..63

  v8f o[4] = {};
  float mrow[8], lrow[8];
#pragma unroll
  for (int r = 0; r < 8; ++r) { mrow[r] = -1e30f; lrow[r] = 0.f; }

  const float scale = 0.125f;           // 1/sqrt(64)

  for (int j0 = 0; j0 < q0 + 16; j0 += 32) {
    v8f s[2];
#pragma unroll
    for (int t = 0; t < 2; ++t) {
      const int n0 = j0 + t * 16;       // wave-uniform
      v8f z = {};
      if (n0 < SSQ) {
        const __bf16* Kp = Kb + (size_t)(b * SSQ + n0) * DM + h * DKH;
        v16bf kb0 = load_frag(Kp, DM);
        v16bf kb1 = load_frag(Kp + 32, DM);
        z = wmma_bf16(qa0, kb0, z);
        z = wmma_bf16(qa1, kb1, z);
      }
#pragma unroll
      for (int r = 0; r < 8; ++r) {
        int key = n0 + col, qr = q0 + rb + r;
        s[t][r] = (n0 < SSQ && key <= qr) ? z[r] * scale : -1e9f;
      }
    }

    // online softmax: rows live in 16-lane halves -> width-16 butterflies
#pragma unroll
    for (int r = 0; r < 8; ++r) {
      float v = fmaxf(s[0][r], s[1][r]);
#pragma unroll
      for (int off = 8; off >= 1; off >>= 1) v = fmaxf(v, __shfl_xor(v, off, 16));
      float mnew  = fmaxf(mrow[r], v);
      float alpha = __expf(mrow[r] - mnew);
      mrow[r] = mnew;
      float p0 = __expf(s[0][r] - mnew);
      float p1 = __expf(s[1][r] - mnew);
      s[0][r] = p0; s[1][r] = p1;
      float rs = p0 + p1;
#pragma unroll
      for (int off = 8; off >= 1; off >>= 1) rs += __shfl_xor(rs, off, 16);
      lrow[r] = lrow[r] * alpha + rs;
#pragma unroll
      for (int t = 0; t < 4; ++t) o[t][r] *= alpha;
    }

    // stage P (16 queries x 32 keys) through LDS to re-enter WMMA A layout
#pragma unroll
    for (int t = 0; t < 2; ++t)
#pragma unroll
      for (int r = 0; r < 8; ++r)
        Plds[(rb + r) * 32 + t * 16 + col] = (__bf16)s[t][r];
    __syncthreads();
    v16bf pa = load_frag(Plds, 32);
    __syncthreads();

    // O += P * V  (V pre-transposed: [bh][d][s], contiguous in s)
    const __bf16* Vp = Vt + (size_t)bh * DKH * SSQ + j0;
#pragma unroll
    for (int t = 0; t < 4; ++t) {
      v16bf vf = load_frag(Vp + (size_t)(t * 16) * SSQ, SSQ);
      o[t] = wmma_bf16(pa, vf, o[t]);
    }
  }

  __bf16* Op = Ob + (size_t)(b * SSQ + q0) * DM + h * DKH;
#pragma unroll
  for (int r = 0; r < 8; ++r) {
    float inv = 1.f / lrow[r];
#pragma unroll
    for (int t = 0; t < 4; ++t)
      Op[(size_t)(rb + r) * DM + t * 16 + col] = (__bf16)(o[t][r] * inv);
  }
}

// ---------------------------------------------------------------------------
// h = LayerNorm(a + c) * g + be ; write f32 and optional bf16 copies.
// ---------------------------------------------------------------------------
__global__ void add_ln(const float* __restrict__ a, const float* __restrict__ c,
                       const float* __restrict__ g, const float* __restrict__ be,
                       float* __restrict__ outF, __bf16* __restrict__ outB) {
  const int tok = blockIdx.x;
  const int tid = threadIdx.x;
  __shared__ float red[256];

  float vals[4];
  float s = 0.f;
#pragma unroll
  for (int i = 0; i < 4; ++i) {
    int idx = tid + i * 256;
    vals[i] = a[(size_t)tok * DM + idx] + c[(size_t)tok * DM + idx];
    s += vals[i];
  }
  red[tid] = s; __syncthreads();
  for (int off = 128; off > 0; off >>= 1) {
    if (tid < off) red[tid] += red[tid + off];
    __syncthreads();
  }
  float mu = red[0] * (1.f / DM);
  __syncthreads();

  float s2 = 0.f;
#pragma unroll
  for (int i = 0; i < 4; ++i) { float d = vals[i] - mu; s2 += d * d; }
  red[tid] = s2; __syncthreads();
  for (int off = 128; off > 0; off >>= 1) {
    if (tid < off) red[tid] += red[tid + off];
    __syncthreads();
  }
  float rstd = rsqrtf(red[0] * (1.f / DM) + 1e-5f);

#pragma unroll
  for (int i = 0; i < 4; ++i) {
    int idx = tid + i * 256;
    float y = (vals[i] - mu) * rstd * g[idx] + be[idx];
    if (outF) outF[(size_t)tok * DM + idx] = y;
    if (outB) outB[(size_t)tok * DM + idx] = (__bf16)y;
  }
}

// ---------------------------------------------------------------------------
extern "C" void kernel_launch(void* const* d_in, const int* in_sizes, int n_in,
                              void* d_out, int out_size, void* d_ws, size_t ws_size,
                              hipStream_t stream) {
  (void)in_sizes; (void)n_in; (void)out_size; (void)ws_size;
  const float* x   = (const float*)d_in[0];
  // d_in[1] = causal mask (structure known; unused)
  const float* wq  = (const float*)d_in[2];  const float* bq  = (const float*)d_in[3];
  const float* wk  = (const float*)d_in[4];  const float* bk  = (const float*)d_in[5];
  const float* wv  = (const float*)d_in[6];  const float* bv  = (const float*)d_in[7];
  const float* wo  = (const float*)d_in[8];  const float* bo  = (const float*)d_in[9];
  const float* w1  = (const float*)d_in[10]; const float* b1  = (const float*)d_in[11];
  const float* w2  = (const float*)d_in[12]; const float* b2  = (const float*)d_in[13];
  const float* g1  = (const float*)d_in[14]; const float* be1 = (const float*)d_in[15];
  const float* g2  = (const float*)d_in[16]; const float* be2 = (const float*)d_in[17];
  float* out = (float*)d_out;

  char* ws = (char*)d_ws;
  size_t off = 0;
  auto alloc = [&](size_t bytes) {
    void* p = ws + off;
    off = (off + bytes + 255) & ~(size_t)255;
    return p;
  };

  __bf16* xb    = (__bf16*)alloc((size_t)MTOK * DM * 2);
  __bf16* wqb   = (__bf16*)alloc((size_t)DM * DM * 2);
  __bf16* wkb   = (__bf16*)alloc((size_t)DM * DM * 2);
  __bf16* wvb   = (__bf16*)alloc((size_t)DM * DM * 2);
  __bf16* wob   = (__bf16*)alloc((size_t)DM * DM * 2);
  __bf16* w1b   = (__bf16*)alloc((size_t)DFF * DM * 2);
  __bf16* w2b   = (__bf16*)alloc((size_t)DM * DFF * 2);
  __bf16* Qb    = (__bf16*)alloc((size_t)MTOK * DM * 2);
  __bf16* Kbuf  = (__bf16*)alloc((size_t)MTOK * DM * 2);
  __bf16* Vbuf  = (__bf16*)alloc((size_t)MTOK * DM * 2);
  __bf16* Vtb   = (__bf16*)alloc((size_t)MTOK * DM * 2);
  __bf16* attnb = (__bf16*)alloc((size_t)MTOK * DM * 2);
  float*  projf = (float*)alloc((size_t)MTOK * DM * 4);
  float*  hf    = (float*)alloc((size_t)MTOK * DM * 4);
  __bf16* hb    = (__bf16*)alloc((size_t)MTOK * DM * 2);
  __bf16* ffb   = (__bf16*)alloc((size_t)MTOK * DFF * 2);
  float*  ff2f  = (float*)alloc((size_t)MTOK * DM * 4);

  // --- convert inputs/weights to bf16 ---
  cvt_f32_bf16<<<4096, 256, 0, stream>>>(x,  xb,  MTOK * DM);
  cvt_f32_bf16<<<2048, 256, 0, stream>>>(wq, wqb, DM * DM);
  cvt_f32_bf16<<<2048, 256, 0, stream>>>(wk, wkb, DM * DM);
  cvt_f32_bf16<<<2048, 256, 0, stream>>>(wv, wvb, DM * DM);
  cvt_f32_bf16<<<2048, 256, 0, stream>>>(wo, wob, DM * DM);
  cvt_f32_bf16<<<4096, 256, 0, stream>>>(w1, w1b, DFF * DM);
  cvt_f32_bf16<<<4096, 256, 0, stream>>>(w2, w2b, DM * DFF);

  auto gemm = [&](const __bf16* A, const __bf16* W, const float* bias,
                  float* oF, __bf16* oB, int Md, int Nd, int Kd, int relu) {
    dim3 grid(Nd / 128, Md / 128);
    gemm_wmma<<<grid, dim3(256), 0, stream>>>(A, W, bias, oF, oB, Nd, Kd, relu);
  };

  // --- QKV projections ---
  gemm(xb, wqb, bq, nullptr, Qb,   MTOK, DM, DM, 0);
  gemm(xb, wkb, bk, nullptr, Kbuf, MTOK, DM, DM, 0);
  gemm(xb, wvb, bv, nullptr, Vbuf, MTOK, DM, DM, 0);

  transpose_v<<<4096, 256, 0, stream>>>(Vbuf, Vtb);

  // --- attention ---
  attn_flash<<<dim3(SSQ / 16, BB * NH), dim3(32), 0, stream>>>(Qb, Kbuf, Vtb, attnb);

  // --- output projection + LN1 ---
  gemm(attnb, wob, bo, projf, nullptr, MTOK, DM, DM, 0);
  add_ln<<<MTOK, 256, 0, stream>>>(x, projf, g1, be1, hf, hb);

  // --- FFN + LN2 ---
  gemm(hb, w1b, b1, nullptr, ffb, MTOK, DFF, DM, 1);
  gemm(ffb, w2b, b2, ff2f, nullptr, MTOK, DM, DFF, 0);
  add_ln<<<MTOK, 256, 0, stream>>>(hf, ff2f, g2, be2, out, nullptr);
}
